// ARCViT_24172075942516
// MI455X (gfx1250) — matrix operand order repaired and checked by reading, not compiled, not run on hardware
//
#include <hip/hip_runtime.h>
#include <hip/hip_bf16.h>
#include <math.h>

// ---------------------------------------------------------------------------
// ARC ViT forward, MI455X (gfx1250, wave32, WMMA).
// All GEMMs use v_wmma_f32_16x16x32_bf16 (bf16 in, fp32 accum).
// GEMM tiles staged with global_load_async_to_lds_b128 (ASYNCcnt),
// double-buffered LDS. Attention: flash-style online softmax per (b,h);
// row sums computed with an extra WMMA against a ones-matrix.
// ---------------------------------------------------------------------------

typedef __attribute__((ext_vector_type(16))) __bf16 v16bf;
typedef __attribute__((ext_vector_type(8)))  __bf16 v8bf;
typedef __attribute__((ext_vector_type(8)))  float  v8f;

#define BATCH 64
#define NHEAD 12
#define SLEN  257     // 1 task token + 256 patches
#define DMODEL 768
#define DMLP  3072
#define HD    64
#define MROWS (BATCH * SLEN)   // 16448
#define SCALE 0.125f           // 64^-0.5

__device__ __forceinline__ v16bf combine16(v8bf lo, v8bf hi) {
  v16bf r;
#pragma unroll
  for (int j = 0; j < 8; ++j) { r[j] = lo[j]; r[8 + j] = hi[j]; }
  return r;
}

__device__ __forceinline__ void async_b128(unsigned lds_off,
                                           const void* gaddr) {
  asm volatile("global_load_async_to_lds_b128 %0, %1, off"
               :: "v"(lds_off), "v"((unsigned long long)(uintptr_t)gaddr)
               : "memory");
}

__device__ __forceinline__ unsigned lds_addr(const void* p) {
  return (unsigned)(uintptr_t)p;   // low 32 bits of generic addr = LDS offset
}

// ---------------------------------------------------------------------------
// Generic tiled GEMM:  out[M,N] = A[M,K] (bf16 rowmajor) * W[N,K]^T + bias
// EPI: 0 = fp32 store, 1 = bf16 store, 2 = exact-GELU -> bf16 store
// Block: 256 threads = 8 waves. Block tile 128x128, wave tile 64x32.
// Double-buffered LDS, async global->LDS staging.
// ---------------------------------------------------------------------------
template <int EPI>
__global__ __launch_bounds__(256) void gemm_kernel(
    const __bf16* __restrict__ A, const __bf16* __restrict__ W,
    const float* __restrict__ bias, void* __restrict__ outv,
    int M, int N, int K)
{
  __shared__ __align__(16) __bf16 As[2][128][40];
  __shared__ __align__(16) __bf16 Ws[2][128][40];

  const int tid  = threadIdx.x;
  const int w    = tid >> 5;
  const int lane = tid & 31;
  const int ln   = lane & 15;
  const bool hiL = lane >= 16;
  const int wm = w >> 2, wn = w & 3;
  const int mBase = blockIdx.y * 128;
  const int nBase = blockIdx.x * 128;

  // per-thread staging coordinates (2 slices each of A-tile and W-tile)
  int srow[2], sc8[2], arow[2], wrow[2];
#pragma unroll
  for (int i = 0; i < 2; ++i) {
    int id = tid + i * 256;
    srow[i] = id >> 2;
    sc8[i]  = (id & 3) * 8;
    int ar = mBase + srow[i]; arow[i] = (ar < M) ? ar : (M - 1);
    int br = nBase + srow[i]; wrow[i] = (br < N) ? br : (N - 1);
    // clamped rows feed only never-stored output rows/cols
  }

  auto stage = [&](int buf, int k0) {
#pragma unroll
    for (int i = 0; i < 2; ++i) {
      async_b128(lds_addr(&As[buf][srow[i]][sc8[i]]),
                 &A[(size_t)arow[i] * K + k0 + sc8[i]]);
      async_b128(lds_addr(&Ws[buf][srow[i]][sc8[i]]),
                 &W[(size_t)wrow[i] * K + k0 + sc8[i]]);
    }
  };

  v8f acc[4][2] = {};
  const int nk = K >> 5;

  stage(0, 0);
  asm volatile("s_wait_asynccnt 0x0" ::: "memory");
  __syncthreads();

  for (int ki = 0; ki < nk; ++ki) {
    const int buf = ki & 1;
    if (ki + 1 < nk) stage(buf ^ 1, (ki + 1) << 5);   // prefetch next tile

    v16bf afrag[4], wfrag[2];
#pragma unroll
    for (int mt = 0; mt < 4; ++mt) {
      int m = wm * 64 + mt * 16 + ln;
      int koff = hiL ? 8 : 0;
      afrag[mt] = combine16(*(const v8bf*)&As[buf][m][koff],
                            *(const v8bf*)&As[buf][m][koff + 16]);
    }
#pragma unroll
    for (int nt = 0; nt < 2; ++nt) {
      int n = wn * 32 + nt * 16 + ln;
      int koff = hiL ? 8 : 0;
      wfrag[nt] = combine16(*(const v8bf*)&Ws[buf][n][koff],
                            *(const v8bf*)&Ws[buf][n][koff + 16]);
    }
#pragma unroll
    for (int mt = 0; mt < 4; ++mt)
#pragma unroll
      for (int nt = 0; nt < 2; ++nt)
        acc[mt][nt] = __builtin_amdgcn_wmma_f32_16x16x32_bf16(
            false, afrag[mt], false, wfrag[nt], (short)0, acc[mt][nt],
            false, false);

    asm volatile("s_wait_asynccnt 0x0" ::: "memory");
    __syncthreads();
  }

  // epilogue
  const int mb = hiL ? 8 : 0;
#pragma unroll
  for (int mt = 0; mt < 4; ++mt) {
#pragma unroll
    for (int nt = 0; nt < 2; ++nt) {
      int col = nBase + wn * 32 + nt * 16 + ln;
#pragma unroll
      for (int r = 0; r < 8; ++r) {
        int row = mBase + wm * 64 + mt * 16 + mb + r;
        if (row < M && col < N) {
          float v = acc[mt][nt][r] + (bias ? bias[col] : 0.f);
          if (EPI == 2) v = 0.5f * v * (1.f + erff(v * 0.70710678118654752f));
          if (EPI == 0) ((float*)outv)[(size_t)row * N + col] = v;
          else ((__bf16*)outv)[(size_t)row * N + col] = (__bf16)v;
        }
      }
    }
  }
}

// ---------------------------------------------------------------------------
// Flash attention per (b,h).  q/k/v bf16 [B*H, S, 64]; ctx bf16 [B, S, 768].
// 8 waves; each wave owns 16 query rows, streams K/V in chunks of 32.
// Row sums of exp(P) computed via WMMA against ones-matrix.
// ---------------------------------------------------------------------------
#define SPAD 296
__global__ __launch_bounds__(256) void attn_kernel(
    const __bf16* __restrict__ qh, const __bf16* __restrict__ kh,
    const __bf16* __restrict__ vh, __bf16* __restrict__ ctx)
{
  __shared__ __align__(16) __bf16 Vt[HD][SPAD];       // V^T staged once
  __shared__ __align__(16) __bf16 Ps[8][16][40];      // per-wave P tile

  const int bh = blockIdx.x;
  const int b = bh / NHEAD, h = bh % NHEAD;
  const int tid = threadIdx.x;
  const int w = tid >> 5, lane = tid & 31;
  const int ln = lane & 15;
  const bool hiL = lane >= 16;

  const __bf16* qb = qh + (size_t)bh * SLEN * HD;
  const __bf16* kb = kh + (size_t)bh * SLEN * HD;
  const __bf16* vb = vh + (size_t)bh * SLEN * HD;

  v16bf ones;
#pragma unroll
  for (int j = 0; j < 16; ++j) ones[j] = (__bf16)1.f;

  // stage V^T into LDS (pad rows beyond S with zeros)
  for (int i = tid; i < HD * 288; i += 256) {
    int d = i & 63, s = i >> 6;
    Vt[d][s] = (s < SLEN) ? vb[s * HD + d] : (__bf16)0.f;
  }
  __syncthreads();

  for (int qt = w; qt < 17; qt += 8) {     // 17 query tiles of 16 rows
    const int qr = qt * 16;

    // Q fragments for K=64 (two 32-wide A fragments), reused all chunks
    v16bf qf[2];
    {
      int row = qr + ln; if (row > SLEN - 1) row = SLEN - 1;
      const __bf16* qrow = qb + row * HD;
#pragma unroll
      for (int ks = 0; ks < 2; ++ks) {
        int d0 = ks * 32 + (hiL ? 8 : 0);
        qf[ks] = combine16(*(const v8bf*)(qrow + d0),
                           *(const v8bf*)(qrow + d0 + 16));
      }
    }

    v8f cacc[4] = {};
    float mrow[8], lrow[8];
#pragma unroll
    for (int r = 0; r < 8; ++r) { mrow[r] = -1e30f; lrow[r] = 0.f; }

    for (int c0 = 0; c0 < 288; c0 += 32) {          // key chunks of 32
      v8f sacc[2] = {};
#pragma unroll
      for (int st = 0; st < 2; ++st) {
        int kcol = c0 + st * 16 + ln;
        int krow = (kcol < SLEN) ? kcol : SLEN - 1;
        const __bf16* krp = kb + krow * HD;
#pragma unroll
        for (int ks = 0; ks < 2; ++ks) {
          int d0 = ks * 32 + (hiL ? 8 : 0);
          v16bf kf = combine16(*(const v8bf*)(krp + d0),
                               *(const v8bf*)(krp + d0 + 16));
          sacc[st] = __builtin_amdgcn_wmma_f32_16x16x32_bf16(
              false, qf[ks], false, kf, (short)0, sacc[st], false, false);
        }
      }

      // scale + mask invalid key columns
      float p[2][8];
#pragma unroll
      for (int st = 0; st < 2; ++st) {
        bool valid = (c0 + st * 16 + ln) < SLEN;
#pragma unroll
        for (int r = 0; r < 8; ++r)
          p[st][r] = valid ? sacc[st][r] * SCALE : -1e30f;
      }

      // online softmax: running max via 16-lane shuffles, exp in registers
      float corr[8];
#pragma unroll
      for (int r = 0; r < 8; ++r) {
        float m = fmaxf(p[0][r], p[1][r]);
#pragma unroll
        for (int o = 1; o < 16; o <<= 1) m = fmaxf(m, __shfl_xor(m, o, 16));
        float mnew = fmaxf(mrow[r], m);
        corr[r] = __expf(mrow[r] - mnew);
        mrow[r] = mnew;
        p[0][r] = __expf(p[0][r] - mnew);
        p[1][r] = __expf(p[1][r] - mnew);
#pragma unroll
        for (int nt = 0; nt < 4; ++nt) cacc[nt][r] *= corr[r];
      }

      // scatter P (C-layout) into per-wave LDS tile, reload as A-fragment
      const int mb = hiL ? 8 : 0;
#pragma unroll
      for (int st = 0; st < 2; ++st)
#pragma unroll
        for (int r = 0; r < 8; ++r)
          Ps[w][mb + r][st * 16 + ln] = (__bf16)p[st][r];
      asm volatile("s_wait_dscnt 0x0" ::: "memory");   // same-wave LDS RAW

      v16bf pf;
      {
        int koff = hiL ? 8 : 0;
        pf = combine16(*(const v8bf*)&Ps[w][ln][koff],
                       *(const v8bf*)&Ps[w][ln][koff + 16]);
      }

      // row sums of exp(P) via matrix pipe: P(16x32) x ones(32x16)
      v8f zsum = {};
      zsum = __builtin_amdgcn_wmma_f32_16x16x32_bf16(
          false, pf, false, ones, (short)0, zsum, false, false);
#pragma unroll
      for (int r = 0; r < 8; ++r) lrow[r] = lrow[r] * corr[r] + zsum[r];

#pragma unroll
      for (int nt = 0; nt < 4; ++nt) {
        int d = nt * 16 + ln;
        int koff = c0 + (hiL ? 8 : 0);
        v16bf vf = combine16(*(const v8bf*)&Vt[d][koff],
                             *(const v8bf*)&Vt[d][koff + 16]);
        cacc[nt] = __builtin_amdgcn_wmma_f32_16x16x32_bf16(
            false, pf, false, vf, (short)0, cacc[nt], false, false);
      }
    }

    // normalize + store ctx at [b, s, h*64+d]
    const int mb = hiL ? 8 : 0;
#pragma unroll
    for (int r = 0; r < 8; ++r) {
      int row = qr + mb + r;
      if (row < SLEN) {
        float inv = 1.f / lrow[r];
#pragma unroll
        for (int nt = 0; nt < 4; ++nt) {
          int col = h * HD + nt * 16 + ln;
          ctx[((size_t)b * SLEN + row) * DMODEL + col] =
              (__bf16)(cacc[nt][r] * inv);
        }
      }
    }
  }
}

// ---------------------------------------------------------------------------
// Residual add + LayerNorm (res may be null). One row per block (D=768).
// ---------------------------------------------------------------------------
__global__ __launch_bounds__(256) void add_ln_kernel(
    const float* __restrict__ x, const float* __restrict__ res,
    const float* __restrict__ g, const float* __restrict__ bta,
    float* __restrict__ xo, __bf16* __restrict__ xb)
{
  const int row = blockIdx.x, t = threadIdx.x;
  __shared__ float red[256];
  float v[3];
#pragma unroll
  for (int i = 0; i < 3; ++i) {
    int c = t + i * 256;
    float val = x[(size_t)row * DMODEL + c];
    if (res) val += res[(size_t)row * DMODEL + c];
    v[i] = val;
  }
  float s = v[0] + v[1] + v[2];
  red[t] = s; __syncthreads();
  for (int o = 128; o > 0; o >>= 1) { if (t < o) red[t] += red[t + o]; __syncthreads(); }
  float mean = red[0] * (1.f / DMODEL);
  __syncthreads();
  float sq = 0.f;
#pragma unroll
  for (int i = 0; i < 3; ++i) { float d = v[i] - mean; sq += d * d; }
  red[t] = sq; __syncthreads();
  for (int o = 128; o > 0; o >>= 1) { if (t < o) red[t] += red[t + o]; __syncthreads(); }
  float rstd = rsqrtf(red[0] * (1.f / DMODEL) + 1e-5f);
#pragma unroll
  for (int i = 0; i < 3; ++i) {
    int c = t + i * 256;
    float o = (v[i] - mean) * rstd * g[c] + bta[c];
    xo[(size_t)row * DMODEL + c] = o;
    xb[(size_t)row * DMODEL + c] = (__bf16)o;
  }
}

// ---------------------------------------------------------------------------
// RoPE on qkv (bf16 [B,S,3*768]) -> head-major q/k/v bf16 [B*H, S, 64]
// ---------------------------------------------------------------------------
__global__ void rope_kernel(const __bf16* __restrict__ qkv,
                            const float* __restrict__ cosb,
                            const float* __restrict__ sinb,
                            __bf16* __restrict__ qh, __bf16* __restrict__ kh,
                            __bf16* __restrict__ vh)
{
  int idx = blockIdx.x * blockDim.x + threadIdx.x;
  if (idx >= BATCH * NHEAD * SLEN * 32) return;
  int p = idx & 31;
  int t = idx >> 5;
  int s = t % SLEN;
  int bh = t / SLEN;
  int h = bh % NHEAD, b = bh / NHEAD;
  int d0 = p * 2, d1 = d0 + 1;
  size_t rb = ((size_t)b * SLEN + s) * (3 * DMODEL);
  float q0 = (float)qkv[rb + h * HD + d0];
  float q1 = (float)qkv[rb + h * HD + d1];
  float k0 = (float)qkv[rb + DMODEL + h * HD + d0];
  float k1 = (float)qkv[rb + DMODEL + h * HD + d1];
  float v0 = (float)qkv[rb + 2 * DMODEL + h * HD + d0];
  float v1 = (float)qkv[rb + 2 * DMODEL + h * HD + d1];
  float oq0 = q0, oq1 = q1, ok0 = k0, ok1 = k1;
  if (s > 0) {
    float c  = cosb[(s - 1) * HD + d0];
    float sn = sinb[(s - 1) * HD + d0];
    oq0 = q0 * c - q1 * sn; oq1 = q1 * c + q0 * sn;
    ok0 = k0 * c - k1 * sn; ok1 = k1 * c + k0 * sn;
  }
  size_t ob = ((size_t)bh * SLEN + s) * HD;
  qh[ob + d0] = (__bf16)oq0; qh[ob + d1] = (__bf16)oq1;
  kh[ob + d0] = (__bf16)ok0; kh[ob + d1] = (__bf16)ok1;
  vh[ob + d0] = (__bf16)v0;  vh[ob + d1] = (__bf16)v1;
}

__global__ void rope_table_kernel(float* __restrict__ cosb,
                                  float* __restrict__ sinb)
{
  int i = blockIdx.x * blockDim.x + threadIdx.x;
  if (i >= 256 * HD) return;
  int d = i & 63, s = i >> 6;
  int t = (d < 32) ? (s >> 4) : (s & 15);
  int j = (d & 31) >> 1;
  float freq = powf(10000.f, -(float)j / 16.f);
  float ang = (float)t * freq;
  cosb[i] = cosf(ang);
  sinb[i] = sinf(ang);
}

// table[pp][c][d] = sum_i color_embed[c][i] * patch_w[d][i][py][px]
__global__ void patch_table_kernel(const float* __restrict__ ce,
                                   const float* __restrict__ pw,
                                   float* __restrict__ table)
{
  int i = blockIdx.x * blockDim.x + threadIdx.x;
  if (i >= 4 * 10 * DMODEL) return;
  int d = i % DMODEL;
  int c = (i / DMODEL) % 10;
  int pp = i / (10 * DMODEL);
  int py = pp >> 1, px = pp & 1;
  float s = 0.f;
  for (int k = 0; k < DMODEL; ++k)
    s += ce[c * DMODEL + k] * pw[(((size_t)d * DMODEL + k) * 2 + py) * 2 + px];
  table[i] = s;
}

__global__ void embed_kernel(const int* __restrict__ pix,
                             const int* __restrict__ task_ids,
                             const float* __restrict__ task_embed,
                             const float* __restrict__ table,
                             const float* __restrict__ patch_b,
                             const float* __restrict__ pos,
                             float* __restrict__ xf, __bf16* __restrict__ xb)
{
  int i = blockIdx.x * blockDim.x + threadIdx.x;
  if (i >= MROWS * DMODEL) return;
  int d = i % DMODEL;
  int r = i / DMODEL;
  int s = r % SLEN, b = r / SLEN;
  float v;
  if (s == 0) {
    v = task_embed[(size_t)task_ids[b] * DMODEL + d];
  } else {
    int sp = s - 1, gy = sp >> 4, gx = sp & 15;
    v = patch_b[d] + pos[(size_t)sp * DMODEL + d];
#pragma unroll
    for (int py = 0; py < 2; ++py)
#pragma unroll
      for (int px = 0; px < 2; ++px) {
        int c = pix[((size_t)b * 32 + (gy * 2 + py)) * 32 + gx * 2 + px];
        v += table[(size_t)((py * 2 + px) * 10 + c) * DMODEL + d];
      }
  }
  xf[i] = v;
  xb[i] = (__bf16)v;
}

__global__ void f2bf_kernel(const float* __restrict__ in,
                            __bf16* __restrict__ out, int n)
{
  int i = blockIdx.x * blockDim.x + threadIdx.x;
  if (i < n) out[i] = (__bf16)in[i];
}

// logits [M,40] -> out [B, NC, 32, 32]
__global__ void head_out_kernel(const float* __restrict__ hed,
                                float* __restrict__ out)
{
  int i = blockIdx.x * blockDim.x + threadIdx.x;
  if (i >= BATCH * 10 * 32 * 32) return;
  int x = i & 31;
  int y = (i >> 5) & 31;
  int c = (i >> 10) % 10;
  int b = i / (10 * 32 * 32);
  int gy = y >> 1, py = y & 1, gx = x >> 1, px = x & 1;
  out[i] = hed[((size_t)b * SLEN + 1 + gy * 16 + gx) * 40 + (py * 2 + px) * 10 + c];
}

// ---------------------------------------------------------------------------
extern "C" void kernel_launch(void* const* d_in, const int* in_sizes, int n_in,
                              void* d_out, int out_size, void* d_ws,
                              size_t ws_size, hipStream_t stream)
{
  (void)in_sizes; (void)n_in; (void)out_size; (void)ws_size;
  const int*   pixel_values = (const int*)d_in[0];
  const int*   task_ids     = (const int*)d_in[1];
  const float* color_embed  = (const float*)d_in[2];
  const float* task_embed   = (const float*)d_in[3];
  const float* patch_w      = (const float*)d_in[4];
  const float* patch_b      = (const float*)d_in[5];
  const float* pos          = (const float*)d_in[6];
  const float* qkv_w  = (const float*)d_in[7];
  const float* qkv_b  = (const float*)d_in[8];
  const float* proj_w = (const float*)d_in[9];
  const float* proj_b = (const float*)d_in[10];
  const float* ln1_g  = (const float*)d_in[11];
  const float* ln1_b  = (const float*)d_in[12];
  const float* fc1_w  = (const float*)d_in[13];
  const float* fc1_b  = (const float*)d_in[14];
  const float* fc2_w  = (const float*)d_in[15];
  const float* fc2_b  = (const float*)d_in[16];
  const float* ln2_g  = (const float*)d_in[17];
  const float* ln2_b  = (const float*)d_in[18];
  const float* norm_g = (const float*)d_in[19];
  const float* norm_b = (const float*)d_in[20];
  const float* head_w = (const float*)d_in[21];
  const float* head_b = (const float*)d_in[22];

  // workspace bump allocator (256B aligned)
  char* ws = (char*)d_ws;
  size_t off = 0;
  auto alloc = [&](size_t bytes) {
    void* p = ws + off;
    off = (off + bytes + 255) & ~(size_t)255;
    return p;
  };
  __bf16* wbuf = (__bf16*)alloc((size_t)DMLP * DMODEL * 2);      // layer weights bf16
  float*  xf   = (float*) alloc((size_t)MROWS * DMODEL * 4);
  __bf16* xb   = (__bf16*)alloc((size_t)MROWS * DMODEL * 2);
  __bf16* qkvb = (__bf16*)alloc((size_t)MROWS * 3 * DMODEL * 2);
  size_t hsz   = (size_t)BATCH * NHEAD * SLEN * HD;
  __bf16* qhb  = (__bf16*)alloc(hsz * 2);
  __bf16* khb  = (__bf16*)alloc(hsz * 2);
  __bf16* vhb  = (__bf16*)alloc(hsz * 2);
  __bf16* ctxb = (__bf16*)alloc((size_t)MROWS * DMODEL * 2);
  float*  ymf  = (float*) alloc((size_t)MROWS * DMODEL * 4);
  __bf16* hb   = (__bf16*)alloc((size_t)MROWS * DMLP * 2);
  float*  cosb = (float*) alloc((size_t)256 * HD * 4);
  float*  sinb = (float*) alloc((size_t)256 * HD * 4);
  float*  tabl = (float*) alloc((size_t)4 * 10 * DMODEL * 4);

  const int T = 256;
  auto blocks = [](long long n, int t) { return (unsigned)((n + t - 1) / t); };

  // ---- setup ----
  rope_table_kernel<<<blocks(256 * HD, T), T, 0, stream>>>(cosb, sinb);
  patch_table_kernel<<<blocks(4 * 10 * DMODEL, T), T, 0, stream>>>(
      color_embed, patch_w, tabl);
  embed_kernel<<<blocks((long long)MROWS * DMODEL, T), T, 0, stream>>>(
      pixel_values, task_ids, task_embed, tabl, patch_b, pos, xf, xb);

  const dim3 gQKV((3 * DMODEL) / 128, (MROWS + 127) / 128);
  const dim3 gD(DMODEL / 128, (MROWS + 127) / 128);
  const dim3 gMLP(DMLP / 128, (MROWS + 127) / 128);
  const dim3 gHEAD(1, (MROWS + 127) / 128);

  for (int l = 0; l < 12; ++l) {
    // qkv
    f2bf_kernel<<<blocks(3 * DMODEL * DMODEL, T), T, 0, stream>>>(
        qkv_w + (size_t)l * 3 * DMODEL * DMODEL, wbuf, 3 * DMODEL * DMODEL);
    gemm_kernel<1><<<gQKV, T, 0, stream>>>(
        xb, wbuf, qkv_b + (size_t)l * 3 * DMODEL, qkvb, MROWS, 3 * DMODEL, DMODEL);
    rope_kernel<<<blocks((long long)BATCH * NHEAD * SLEN * 32, T), T, 0, stream>>>(
        qkvb, cosb, sinb, qhb, khb, vhb);
    // attention
    attn_kernel<<<BATCH * NHEAD, T, 0, stream>>>(qhb, khb, vhb, ctxb);
    // proj
    f2bf_kernel<<<blocks(DMODEL * DMODEL, T), T, 0, stream>>>(
        proj_w + (size_t)l * DMODEL * DMODEL, wbuf, DMODEL * DMODEL);
    gemm_kernel<0><<<gD, T, 0, stream>>>(
        ctxb, wbuf, proj_b + (size_t)l * DMODEL, ymf, MROWS, DMODEL, DMODEL);
    add_ln_kernel<<<MROWS, T, 0, stream>>>(
        xf, ymf, ln1_g + (size_t)l * DMODEL, ln1_b + (size_t)l * DMODEL, xf, xb);
    // fc1 (+GELU)
    f2bf_kernel<<<blocks(DMLP * DMODEL, T), T, 0, stream>>>(
        fc1_w + (size_t)l * DMLP * DMODEL, wbuf, DMLP * DMODEL);
    gemm_kernel<2><<<gMLP, T, 0, stream>>>(
        xb, wbuf, fc1_b + (size_t)l * DMLP, hb, MROWS, DMLP, DMODEL);
    // fc2
    f2bf_kernel<<<blocks(DMODEL * DMLP, T), T, 0, stream>>>(
        fc2_w + (size_t)l * DMODEL * DMLP, wbuf, DMODEL * DMLP);
    gemm_kernel<0><<<gD, T, 0, stream>>>(
        hb, wbuf, fc2_b + (size_t)l * DMODEL, ymf, MROWS, DMODEL, DMLP);
    add_ln_kernel<<<MROWS, T, 0, stream>>>(
        xf, ymf, ln2_g + (size_t)l * DMODEL, ln2_b + (size_t)l * DMODEL, xf, xb);
  }

  // final norm + head
  add_ln_kernel<<<MROWS, T, 0, stream>>>(xf, nullptr, norm_g, norm_b, xf, xb);
  f2bf_kernel<<<blocks(40 * DMODEL, T), T, 0, stream>>>(head_w, wbuf, 40 * DMODEL);
  gemm_kernel<0><<<gHEAD, T, 0, stream>>>(xb, wbuf, head_b, ymf, MROWS, 40, DMODEL);
  head_out_kernel<<<blocks((long long)BATCH * 10 * 32 * 32, T), T, 0, stream>>>(
      ymf, (float*)d_out);
}
